// MultiInputLSTMCell_76605036691765
// MI455X (gfx1250) — matrix operand order repaired and checked by reading, compiled, not run on hardware
//
#include <hip/hip_runtime.h>
#include <hip/hip_bf16.h>

// ---------------------------------------------------------------------------
// MultiInputLSTMCell for MI455X (gfx1250, wave32, WMMA bf16 path)
//   B=4096, D=512, H=512, C=8
// Pipeline:
//   1) fp32 -> bf16 converts (weights transposed to N-major so WMMA
//      B-fragments are contiguous 2x16B loads per lane)
//   2) padding-mask reduction (wave32 ballot)
//   3) v_wmma_f32_16x16x32_bf16 GEMMs, 32x32 tile per wave (2 A-frags x
//      2 B-frags -> 4 WMMA per K-step), double-buffered fragments for
//      load/WMMA overlap; gates fuses both K=512 loops into one accumulator;
//      alpha_wh batched over C via blockIdx.z
//   4) elementwise sigmoid/tanh/exp-normalize merge -> h_1, c_1
// ---------------------------------------------------------------------------

typedef __attribute__((ext_vector_type(16))) __bf16 v16bf;
typedef __attribute__((ext_vector_type(8)))  __bf16 v8bf;
typedef __attribute__((ext_vector_type(8)))  float  v8f;

constexpr int Bn = 4096;
constexpr int Dn = 512;
constexpr int Hn = 512;
constexpr int Cn = 8;
constexpr float NEG_BIG = -1000000.0f;

// ------------------------- conversion kernels ------------------------------

__global__ void cvt_kernel(const float* __restrict__ in, __bf16* __restrict__ out, int n) {
  int i = blockIdx.x * blockDim.x + threadIdx.x;
  if (i < n) out[i] = (__bf16)in[i];
}

// W is (K x N) row-major fp32; WT becomes (N x K) row-major bf16 (= W^T)
__global__ void cvtT_kernel(const float* __restrict__ W, __bf16* __restrict__ WT, int K, int N) {
  int i = blockIdx.x * blockDim.x + threadIdx.x;
  if (i < K * N) {
    int k = i / N;
    int n = i - k * N;
    WT[(size_t)n * K + k] = (__bf16)W[i];
  }
}

// ------------------------- padding mask ------------------------------------
// One wave32 per (b,c) row of c_input_var; mask = NEG_BIG if the whole H=512
// row is zero, else 1.0.  Rows = B*C = 32768.

__global__ __launch_bounds__(256) void mask_kernel(const float* __restrict__ cin,
                                                   float* __restrict__ mask) {
  const int lane = threadIdx.x & 31;
  const int row  = blockIdx.x * (blockDim.x >> 5) + (threadIdx.x >> 5); // b*C + c
  const float4* p = (const float4*)(cin + (size_t)row * Hn);
  int nz = 0;
  #pragma unroll
  for (int t = 0; t < (Hn / 4) / 32; ++t) {            // 4 iterations
    float4 v = p[lane + t * 32];
    nz |= (v.x != 0.f) | (v.y != 0.f) | (v.z != 0.f) | (v.w != 0.f);
  }
  unsigned long long bal = __ballot(nz);
  if (lane == 0) mask[row] = (bal == 0ull) ? NEG_BIG : 1.0f;
}

// ------------------------- WMMA GEMM ---------------------------------------
// A-fragment (16x32 bf16): lane l (r=l&15, hi=l>>4) holds K = hi*8+{0..7} in
// VGPR0-3 and K = 16+hi*8+{0..7} in VGPR4-7  ->  two 16B loads at +0 and +32B.
// B-fragment (32x16 bf16) with B stored transposed (N-major, row stride K):
// lane l (n=l&15, hi=l>>4) holds K = hi*8+{0..7} and 16+hi*8+{0..7} of
// column n  ->  two 16B loads at +0 and +32B.  C/D: VGPR e, lane l -> row e+8*hi, col n.

__device__ __forceinline__ v16bf load_a_frag(const __bf16* p) {
  union { v16bf v; v8bf h[2]; } u;
  u.h[0] = *(const v8bf*)(p);
  u.h[1] = *(const v8bf*)(p + 16);
  return u.v;
}

__device__ __forceinline__ v16bf load_b_frag(const __bf16* p) {
  union { v16bf v; v8bf h[2]; } u;
  u.h[0] = *(const v8bf*)(p);
  u.h[1] = *(const v8bf*)(p + 16);
  return u.v;
}

struct Frags {
  v16bf a0, a1, b0, b1;
};

__device__ __forceinline__ void load_frags(Frags& f,
                                           const __bf16* __restrict__ arow0,
                                           const __bf16* __restrict__ arow1,
                                           const __bf16* __restrict__ bcol0,
                                           const __bf16* __restrict__ bcol1,
                                           int kk, int hi) {
  f.a0 = load_a_frag(arow0 + kk + hi * 8);
  f.a1 = load_a_frag(arow1 + kk + hi * 8);
  f.b0 = load_b_frag(bcol0 + kk + hi * 8);
  f.b1 = load_b_frag(bcol1 + kk + hi * 8);
}

__device__ __forceinline__ void mma_step(const Frags& f, v8f acc[4]) {
  acc[0] = __builtin_amdgcn_wmma_f32_16x16x32_bf16(false, f.a0, false, f.b0,
                                                   (short)0, acc[0], false, false);
  acc[1] = __builtin_amdgcn_wmma_f32_16x16x32_bf16(false, f.a0, false, f.b1,
                                                   (short)0, acc[1], false, false);
  acc[2] = __builtin_amdgcn_wmma_f32_16x16x32_bf16(false, f.a1, false, f.b0,
                                                   (short)0, acc[2], false, false);
  acc[3] = __builtin_amdgcn_wmma_f32_16x16x32_bf16(false, f.a1, false, f.b1,
                                                   (short)0, acc[3], false, false);
}

// K must be a multiple of 64 (all K here are 512).  Double-buffered
// software pipeline: fragments for step k+1 are in flight while the four
// WMMAs for step k execute -> no full s_wait_loadcnt 0 in steady state.
__device__ __forceinline__ void gemm_loop(const __bf16* __restrict__ arow0,
                                          const __bf16* __restrict__ bcol0,
                                          int lda, int K, int hi, v8f acc[4]) {
  const __bf16* arow1 = arow0 + (size_t)16 * lda;
  const __bf16* bcol1 = bcol0 + (size_t)16 * K;

  Frags f0, f1;
  load_frags(f0, arow0, arow1, bcol0, bcol1, 0, hi);
  int kk = 32;
  for (int it = 0; it < K / 64 - 1; ++it, kk += 64) {
    load_frags(f1, arow0, arow1, bcol0, bcol1, kk, hi);
    __builtin_prefetch(arow0 + kk + 96, 0, 3);   // global_prefetch_b8
    mma_step(f0, acc);
    load_frags(f0, arow0, arow1, bcol0, bcol1, kk + 32, hi);
    mma_step(f1, acc);
  }
  load_frags(f1, arow0, arow1, bcol0, bcol1, kk, hi);
  mma_step(f0, acc);
  mma_step(f1, acc);
}

// Cout[M x N] (fp32, ldc) = A0[M x K0] * BT0^T + (A1 ? A1[M x K1] * BT1^T : 0) + bias
// blockIdx.z batches: A0 += z*aBatch elems, Cout += z*cBatch elems.
// Block = 128 threads = 4 waves; each wave owns a 32x32 output tile
// (waves stacked in M) -> block tile 128x32.
__global__ __launch_bounds__(128) void wmma_gemm_kernel(
    const __bf16* __restrict__ A0, int lda0, const __bf16* __restrict__ BT0, int K0,
    const __bf16* __restrict__ A1, int lda1, const __bf16* __restrict__ BT1, int K1,
    const float* __restrict__ bias, float* __restrict__ Cout, int ldc,
    long long aBatch, long long cBatch) {
  const int lane = threadIdx.x & 31;
  const int wave = threadIdx.x >> 5;
  const int lo = lane & 15;
  const int hi = lane >> 4;
  const int m0 = (blockIdx.y * 4 + wave) * 32;   // 32 rows per wave
  const int n0 = blockIdx.x * 32;                // 2 x 16 cols per wave
  const int z  = blockIdx.z;

  A0   += (size_t)z * aBatch;
  Cout += (size_t)z * cBatch;

  v8f acc[4];
  const v8f vzero = {0.f, 0.f, 0.f, 0.f, 0.f, 0.f, 0.f, 0.f};
  #pragma unroll
  for (int t = 0; t < 4; ++t) acc[t] = vzero;

  gemm_loop(A0 + (size_t)(m0 + lo) * lda0, BT0 + (size_t)(n0 + lo) * K0,
            lda0, K0, hi, acc);
  if (A1)  // fused second GEMM into the same accumulators (gates = Wi*x + Wh*h)
    gemm_loop(A1 + (size_t)(m0 + lo) * lda1, BT1 + (size_t)(n0 + lo) * K1,
              lda1, K1, hi, acc);

  // acc[ai*2+bi]: rows m0 + ai*16 + e + 8*hi, col n0 + bi*16 + lo
  #pragma unroll
  for (int ai = 0; ai < 2; ++ai) {
    #pragma unroll
    for (int bi = 0; bi < 2; ++bi) {
      const int col = n0 + bi * 16 + lo;
      const float bv = bias ? bias[col] : 0.f;
      const v8f& a = acc[ai * 2 + bi];
      #pragma unroll
      for (int e = 0; e < 8; ++e) {
        const int row = m0 + ai * 16 + e + hi * 8;
        Cout[(size_t)row * ldc + col] = a[e] + bv;
      }
    }
  }
}

// ------------------------- elementwise merge -------------------------------

__device__ __forceinline__ float sigmoidf_(float x) { return 1.0f / (1.0f + __expf(-x)); }

__global__ __launch_bounds__(256) void fuse_kernel(
    const float* __restrict__ gates,    // (B, 3H)
    const float* __restrict__ aWI,      // (B, H)
    const float* __restrict__ aWH,      // (C, B, H)
    const float* __restrict__ cin,      // (B, C, H) fp32 original
    const float* __restrict__ mask,     // (B, C)
    float* __restrict__ out) {          // h_1 (B,H) || c_1 (B,H)
  const int idx = blockIdx.x * blockDim.x + threadIdx.x;  // b*H + h
  const int b = idx >> 9;               // /H
  const int h = idx & (Hn - 1);

  const size_t grow = (size_t)b * (3 * Hn);
  const float i_ = sigmoidf_(gates[grow + h]);
  const float o_ = sigmoidf_(gates[grow + Hn + h]);
  const float g_ = tanhf(gates[grow + 2 * Hn + h]);

  const float awi = aWI[idx];
  float ei = __expf(i_);
  float esum = ei;
  float accv = g_ * ei;
  #pragma unroll
  for (int c = 0; c < Cn; ++c) {
    float a = sigmoidf_(awi + aWH[(size_t)c * Bn * Hn + idx]) * mask[b * Cn + c];
    float ea = __expf(a);
    esum += ea;
    accv += cin[((size_t)b * Cn + c) * Hn + h] * ea;
  }
  const float c1 = accv / esum;
  out[idx] = o_ * tanhf(c1);
  out[(size_t)Bn * Hn + idx] = c1;
}

// ------------------------- launch ------------------------------------------

extern "C" void kernel_launch(void* const* d_in, const int* in_sizes, int n_in,
                              void* d_out, int out_size, void* d_ws, size_t ws_size,
                              hipStream_t stream) {
  const float* input_ = (const float*)d_in[0];   // (B, D)
  const float* h0     = (const float*)d_in[1];   // (B, H)
  /* d_in[2] = c_0: unused by the reference */
  const float* cin    = (const float*)d_in[3];   // (B, C, H)
  const float* wih    = (const float*)d_in[4];   // (D, 3H)
  const float* whh    = (const float*)d_in[5];   // (H, 3H)
  const float* bias   = (const float*)d_in[6];   // (3H,)
  const float* awih   = (const float*)d_in[7];   // (D, H)
  const float* awhh   = (const float*)d_in[8];   // (H, H)
  const float* abias  = (const float*)d_in[9];   // (H,)
  float* out = (float*)d_out;

  char* ws = (char*)d_ws;
  size_t off = 0;
  auto carve = [&](size_t bytes) -> void* {
    void* p = ws + off;
    off = (off + bytes + 255) & ~(size_t)255;
    return p;
  };

  __bf16* inputBF = (__bf16*)carve((size_t)Bn * Dn * 2);
  __bf16* h0BF    = (__bf16*)carve((size_t)Bn * Hn * 2);
  __bf16* cinBF   = (__bf16*)carve((size_t)Bn * Cn * Hn * 2);
  __bf16* wihT    = (__bf16*)carve((size_t)3 * Hn * Dn * 2);
  __bf16* whhT    = (__bf16*)carve((size_t)3 * Hn * Hn * 2);
  __bf16* awihT   = (__bf16*)carve((size_t)Hn * Dn * 2);
  __bf16* awhhT   = (__bf16*)carve((size_t)Hn * Hn * 2);
  float* gatesWS  = (float*)carve((size_t)Bn * 3 * Hn * 4);
  float* aWI      = (float*)carve((size_t)Bn * Hn * 4);
  float* aWH      = (float*)carve((size_t)Cn * Bn * Hn * 4);
  float* maskWS   = (float*)carve((size_t)Bn * Cn * 4);

  const int TPB = 256;
  // 1) converts (activations straight; weights transposed to N-major)
  cvt_kernel<<<(Bn * Dn) / TPB, TPB, 0, stream>>>(input_, inputBF, Bn * Dn);
  cvt_kernel<<<(Bn * Hn) / TPB, TPB, 0, stream>>>(h0, h0BF, Bn * Hn);
  cvt_kernel<<<(Bn * Cn * Hn) / TPB, TPB, 0, stream>>>(cin, cinBF, Bn * Cn * Hn);
  cvtT_kernel<<<(Dn * 3 * Hn) / TPB, TPB, 0, stream>>>(wih, wihT, Dn, 3 * Hn);
  cvtT_kernel<<<(Hn * 3 * Hn) / TPB, TPB, 0, stream>>>(whh, whhT, Hn, 3 * Hn);
  cvtT_kernel<<<(Dn * Hn) / TPB, TPB, 0, stream>>>(awih, awihT, Dn, Hn);
  cvtT_kernel<<<(Hn * Hn) / TPB, TPB, 0, stream>>>(awhh, awhhT, Hn, Hn);

  // 2) padding mask: one wave per (b,c) row, 8 waves per block
  mask_kernel<<<(Bn * Cn) / 8, 256, 0, stream>>>(cin, maskWS);

  // 3) WMMA GEMMs (block tile 128x32, wave tile 32x32)
  // gates = input*W_ih + h0*W_hh + bias : M=4096, N=1536, dual K=512
  wmma_gemm_kernel<<<dim3(3 * Hn / 32, Bn / 128, 1), 128, 0, stream>>>(
      inputBF, Dn, wihT, Dn, h0BF, Hn, whhT, Hn, bias, gatesWS, 3 * Hn, 0, 0);
  // alpha_wi = input*alpha_W_ih + alpha_bias : M=4096, N=512, K=512
  wmma_gemm_kernel<<<dim3(Hn / 32, Bn / 128, 1), 128, 0, stream>>>(
      inputBF, Dn, awihT, Dn, nullptr, 0, nullptr, 0, abias, aWI, Hn, 0, 0);
  // alpha_wh[c] = c_input_var[:,c,:]*alpha_W_hh : 8 batched GEMMs via z
  wmma_gemm_kernel<<<dim3(Hn / 32, Bn / 128, Cn), 128, 0, stream>>>(
      cinBF, Cn * Hn, awhhT, Hn, nullptr, 0, nullptr, 0, nullptr, aWH, Hn,
      (long long)Hn, (long long)Bn * Hn);

  // 4) elementwise merge -> h_1 || c_1
  fuse_kernel<<<(Bn * Hn) / TPB, TPB, 0, stream>>>(gatesWS, aWI, aWH, cin, maskWS, out);
}
